// DripHead_52278341927058
// MI455X (gfx1250) — compile-verified, matrix operand order checked by reading, not run on hardware
//
#include <hip/hip_runtime.h>
#include <hip/hip_bf16.h>

// ---------------------------------------------------------------------------
// Problem constants
// ---------------------------------------------------------------------------
#define B   32
#define S   2048
#define D   1024        // d_model
#define D2  2048        // 2*D (GEMM K for gates)
#define DC  512         // d_model/2
#define V   256         // vocab
#define EPS 1e-6f

#define NBLK_GRU 32     // persistent grid size (must be co-resident)

typedef __attribute__((ext_vector_type(16))) __bf16 v16bf;
typedef __attribute__((ext_vector_type(8)))  __bf16 v8bf;
typedef __attribute__((ext_vector_type(8)))  float  v8f;
typedef __attribute__((ext_vector_type(4)))  unsigned int v4u;
typedef __attribute__((ext_vector_type(8)))  int v8i;
typedef __attribute__((ext_vector_type(4)))  int v4i;

#if defined(__HIP_DEVICE_COMPILE__) && defined(__has_builtin)
#  if __has_builtin(__builtin_amdgcn_tensor_load_to_lds)
#    define HAVE_TDM 1
#  endif
#endif
#ifndef HAVE_TDM
#  define HAVE_TDM 0
#endif

// ---------------------------------------------------------------------------
// Workspace layout (bytes)
// ---------------------------------------------------------------------------
static constexpr size_t WS_BAR = 0;                       // 64 B  (grid barrier cnt/gen)
static constexpr size_t WS_H   = 64;                      // 2 * 32*1024*4 (ping-pong h, fp32)
static constexpr size_t WS_R   = WS_H + 2ull * B * D * 4; // sigmoid r
static constexpr size_t WS_U   = WS_R + (size_t)B * D * 4;// sigmoid u
static constexpr size_t WS_WP  = WS_U + (size_t)B * D * 4;// 12 MB packed gate W (bf16)
static constexpr size_t WS_WO  = WS_WP + 3ull * 64 * 64 * 32 * 16 * 2; // 0.5 MB packed out W
static constexpr size_t WS_HID = WS_WO + 2ull * 16 * 16 * 32 * 16 * 2; // 256 MB hidden states fp32
static constexpr size_t WS_XS  = WS_HID + (size_t)B * S * D * 4;       // 128 MB embedded seq bf16

// d_out layout (floats): fc_r, fc_i, n_r, n_i, logits
static constexpr size_t O_FCR = 0;
static constexpr size_t O_FCI = O_FCR + (size_t)B * DC;
static constexpr size_t O_NR  = O_FCI + (size_t)B * DC;
static constexpr size_t O_NI  = O_NR + (size_t)B * S * DC;
static constexpr size_t O_LOG = O_NI + (size_t)B * S * DC;

// ---------------------------------------------------------------------------
// Helpers
// ---------------------------------------------------------------------------
__device__ __forceinline__ v16bf cat8(v8bf lo, v8bf hi) {
    v16bf r;
#pragma unroll
    for (int i = 0; i < 8; ++i) { r[i] = lo[i]; r[8 + i] = hi[i]; }
    return r;
}

__device__ __forceinline__ void grid_barrier(unsigned* cnt, unsigned* gen, unsigned nblk) {
    __syncthreads();
    if (threadIdx.x == 0) {
        __threadfence();
        unsigned g = __atomic_load_n(gen, __ATOMIC_ACQUIRE);
        if (atomicAdd(cnt, 1u) == nblk - 1u) {
            __atomic_store_n(cnt, 0u, __ATOMIC_RELAXED);
            __atomic_fetch_add(gen, 1u, __ATOMIC_RELEASE);
        } else {
            while (__atomic_load_n(gen, __ATOMIC_ACQUIRE) == g) {
                __builtin_amdgcn_s_sleep(8);
            }
        }
        __threadfence();
    }
    __syncthreads();
}

// fast activations: single v_exp_f32 + v_rcp_f32 instead of precise-div chains
__device__ __forceinline__ float fast_sigmoid(float x) {
    return __builtin_amdgcn_rcpf(1.0f + __expf(-x));
}
__device__ __forceinline__ float fast_tanh(float x) {
    float e = __expf(2.0f * x);
    return 1.0f - 2.0f * __builtin_amdgcn_rcpf(e + 1.0f);
}

#if HAVE_TDM
// Issue a 2-D TDM load: 32 rows x 1024 bf16, row stride S*D elements,
// from xs + t*D elements into LDS at byte offset lds_off. (D# per ISA ch.8)
__device__ __forceinline__ void tdm_load_xt(const __bf16* xs, int t, unsigned lds_off) {
    unsigned long long ga = (unsigned long long)(size_t)xs + (unsigned long long)t * (D * 2);
    v4u g0;
    g0[0] = 1u;                                       // count=1, user descriptor
    g0[1] = lds_off;                                  // LDS byte address of tile
    g0[2] = (unsigned)ga;                             // global_addr[31:0]
    g0[3] = (unsigned)(ga >> 32) | (2u << 30);        // global_addr[56:32] | type=2
    v8i g1;
    g1[0] = (1 << 16);                                // data_size=1 (2 bytes/elem)
    g1[1] = 0;                                        // abar=0, tensor_dim0[15:0]=0
    g1[2] = (int)((((unsigned long long)S * D) >> 16) & 0xFFFFu) | (B << 16); // dim0 hi | tensor_dim1=32
    g1[3] = (D << 16);                                // tile_dim0=1024
    g1[4] = B;                                        // tile_dim1=32, tile_dim2=0
    g1[5] = (int)(S * D);                             // tensor_dim0_stride lo32
    g1[6] = 0;                                        // stride hi, dim1_stride lo
    g1[7] = 0;                                        // dim1_stride hi
    v4i z4 = {0, 0, 0, 0};
    v8i z8 = {0, 0, 0, 0, 0, 0, 0, 0};
    // clang-23 / amdgpu-toolchain 6-arg form:
    // (uint32x4 g0, int32x8 g1, int32x4 g2, int32x4 g3, int32x8 extra, i32 cpol)
    __builtin_amdgcn_tensor_load_to_lds(g0, g1, z4, z4, z8, 0);
}
#endif

// ---------------------------------------------------------------------------
// Kernel 0: init barrier + h state
// ---------------------------------------------------------------------------
__global__ void init_kernel(unsigned* bar, float* h) {
    int i = blockIdx.x * 256 + threadIdx.x;
    if (i < 16) bar[i] = 0u;
    if (i < 2 * B * D) h[i] = 0.0f;
}

// ---------------------------------------------------------------------------
// Kernel 1: embed tokens -> xs bf16 [B,S,D]
// ---------------------------------------------------------------------------
__global__ void embed_kernel(const int* __restrict__ tok,
                             const float* __restrict__ Er,
                             const float* __restrict__ Ei,
                             __bf16* __restrict__ xs) {
    size_t i = (size_t)blockIdx.x * 256 + threadIdx.x;
    if (i >= (size_t)B * S * D) return;
    int d = (int)(i & (D - 1));
    size_t bs = i >> 10;                  // b*S + t
    int tk = tok[bs];
    float v = (d < DC) ? Er[(size_t)tk * DC + d] : Ei[(size_t)tk * DC + (d - DC)];
    xs[i] = (__bf16)v;
}

// ---------------------------------------------------------------------------
// Kernel 2: pack gate weights [2048,1024] fp32 -> bf16 WMMA B-fragment layout
//   wp[(((gate*64+nt)*64+ks)*32+lane)*16 + j]
//   lane<16:  N=lane,    K = ks*32 + {0..7, 16..23}
//   lane>=16: N=lane-16, K = ks*32 + {8..15, 24..31}
// ---------------------------------------------------------------------------
__global__ void pack_gates_kernel(const float* __restrict__ Wr,
                                  const float* __restrict__ Wu,
                                  const float* __restrict__ Wc,
                                  __bf16* __restrict__ wp) {
    int lin = blockIdx.x * 256 + threadIdx.x;
    if (lin >= 3 * 64 * 64 * 32) return;
    int lane = lin & 31;
    int ks   = (lin >> 5) & 63;
    int nt   = (lin >> 11) & 63;
    int gate = lin >> 17;
    const float* W = (gate == 0) ? Wr : ((gate == 1) ? Wu : Wc);
    __bf16* dst = wp + (size_t)lin * 16;
    int n  = nt * 16 + (lane & 15);
    int kb = ks * 32;
    int lo = (lane < 16) ? 0 : 8;
#pragma unroll
    for (int j = 0; j < 8; ++j) dst[j]     = (__bf16)W[(kb + lo + j) * D + n];
#pragma unroll
    for (int j = 0; j < 8; ++j) dst[8 + j] = (__bf16)W[(kb + lo + 16 + j) * D + n];
}

// ---------------------------------------------------------------------------
// Kernel 3: pack output-proj weights [512,256] fp32 -> bf16 B-fragment layout
// ---------------------------------------------------------------------------
__global__ void pack_out_kernel(const float* __restrict__ Wrc,
                                const float* __restrict__ Wic,
                                __bf16* __restrict__ wo) {
    int lin = blockIdx.x * 256 + threadIdx.x;
    if (lin >= 2 * 16 * 16 * 32) return;
    int lane = lin & 31;
    int ks   = (lin >> 5) & 15;
    int nt   = (lin >> 9) & 15;
    int mat  = lin >> 13;
    const float* W = (mat == 0) ? Wrc : Wic;
    __bf16* dst = wo + (size_t)lin * 16;
    int n  = nt * 16 + (lane & 15);
    int kb = ks * 32;
    int lo = (lane < 16) ? 0 : 8;
#pragma unroll
    for (int j = 0; j < 8; ++j) dst[j]     = (__bf16)W[(kb + lo + j) * V + n];
#pragma unroll
    for (int j = 0; j < 8; ++j) dst[8 + j] = (__bf16)W[(kb + lo + 16 + j) * V + n];
}

// ---------------------------------------------------------------------------
// Kernel 4: persistent GRU scan. 32 WGs x 256 threads (8 waves).
//   x-tile for step t+1 is TDM-prefetched (double-buffered) while step t runs.
//   Phase 1: 256 waves -> r|u pre-acts, sigmoid, store.   -> grid barrier
//   Phase 2: 128 waves -> c with A=[xt | r*h], tanh, blend, write h_next
//            and hidden[:,t,:].                           -> grid barrier
// ---------------------------------------------------------------------------
__global__ __launch_bounds__(256) void gru_kernel(
    const __bf16* __restrict__ xs,    // [B,S,D] bf16 embedded inputs
    const float* __restrict__ br, const float* __restrict__ bu,
    const float* __restrict__ bc,
    const __bf16* __restrict__ wp,
    float* __restrict__ h0buf,        // [2][B*D] ping-pong
    float* __restrict__ rbuf, float* __restrict__ ubuf,
    float* __restrict__ hidden,       // [B,S,D]
    unsigned* __restrict__ bar,
    float* __restrict__ out) {

    __shared__ __bf16 xpart[2][B * D]; // 2 x 64 KB  (x tiles, TDM double buffer)
    __shared__ __bf16 hpart[B * D];    // 64 KB      (h or r*h tile)

    const int tid  = threadIdx.x;
    const int lane = tid & 31;
    const int wid  = tid >> 5;
    const int w    = blockIdx.x * 8 + wid;

    unsigned* cnt = bar;
    unsigned* gen = bar + 1;

#if HAVE_TDM
    if (tid == 0) {
        tdm_load_xt(xs, 0, (unsigned)(size_t)&xpart[0][0]);
    }
#endif

    for (int t = 0; t < S; ++t) {
        float* hcur = h0buf + (size_t)(t & 1) * (B * D);
        float* hnxt = h0buf + (size_t)((t + 1) & 1) * (B * D);
        const __bf16* xp = xpart[t & 1];

#if HAVE_TDM
        if (tid == 0) {
            __builtin_amdgcn_s_wait_tensorcnt(0);            // xt(t) in LDS
            if (t + 1 < S)
                tdm_load_xt(xs, t + 1, (unsigned)(size_t)&xpart[(t + 1) & 1][0]);
        }
#else
        for (int i = tid; i < B * D; i += 256) {
            int b = i >> 10, d = i & (D - 1);
            xpart[t & 1][i] = xs[((size_t)b * S + t) * D + d];
        }
#endif
        // ---- stage h into LDS as bf16 ----
        for (int i = tid; i < B * D; i += 256)
            hpart[i] = (__bf16)hcur[i];
        __syncthreads();

        // ---- phase 1: r (gate 0) and u (gate 1) ----
        {
            int gate = w >> 7, rem = w & 127, mt = rem >> 6, nt = rem & 63;
            const __bf16* bp = wp + (size_t)(gate * 64 + nt) * (64 * 32 * 16);
            const int arow = (mt * 16 + (lane & 15)) * D;
            const int kofs = (lane < 16) ? 0 : 8;
            v8f acc = {};
#pragma unroll 4
            for (int ks = 0; ks < 32; ++ks) {                // K half 1: x
                v16bf A = cat8(*(const v8bf*)&xp[arow + ks * 32 + kofs],
                               *(const v8bf*)&xp[arow + ks * 32 + kofs + 16]);
                v16bf Bm = *(const v16bf*)(bp + ((size_t)ks * 32 + lane) * 16);
                acc = __builtin_amdgcn_wmma_f32_16x16x32_bf16(
                    false, A, false, Bm, (short)0, acc, false, false);
            }
#pragma unroll 4
            for (int ks = 0; ks < 32; ++ks) {                // K half 2: h
                v16bf A = cat8(*(const v8bf*)&hpart[arow + ks * 32 + kofs],
                               *(const v8bf*)&hpart[arow + ks * 32 + kofs + 16]);
                v16bf Bm = *(const v16bf*)(bp + ((size_t)(32 + ks) * 32 + lane) * 16);
                acc = __builtin_amdgcn_wmma_f32_16x16x32_bf16(
                    false, A, false, Bm, (short)0, acc, false, false);
            }
            float* dst = (gate == 0) ? rbuf : ubuf;
            const float* bias = (gate == 0) ? br : bu;
            int n = nt * 16 + (lane & 15);
            int mb = mt * 16 + ((lane < 16) ? 0 : 8);
            float bv = bias[n];
#pragma unroll
            for (int v = 0; v < 8; ++v)
                dst[(mb + v) * D + n] = fast_sigmoid(acc[v] + bv);
        }
        grid_barrier(cnt, gen, NBLK_GRU);

        // ---- rebuild h-tile of LDS as r*h ----
        for (int i = tid; i < B * D; i += 256)
            hpart[i] = (__bf16)(rbuf[i] * hcur[i]);
        __syncthreads();

        // ---- phase 2: c (gate 2), blend, write h_next & hidden ----
        if (w < 128) {
            int mt = w >> 6, nt = w & 63;
            const __bf16* bp = wp + (size_t)(2 * 64 + nt) * (64 * 32 * 16);
            const int arow = (mt * 16 + (lane & 15)) * D;
            const int kofs = (lane < 16) ? 0 : 8;
            v8f acc = {};
#pragma unroll 4
            for (int ks = 0; ks < 32; ++ks) {
                v16bf A = cat8(*(const v8bf*)&xp[arow + ks * 32 + kofs],
                               *(const v8bf*)&xp[arow + ks * 32 + kofs + 16]);
                v16bf Bm = *(const v16bf*)(bp + ((size_t)ks * 32 + lane) * 16);
                acc = __builtin_amdgcn_wmma_f32_16x16x32_bf16(
                    false, A, false, Bm, (short)0, acc, false, false);
            }
#pragma unroll 4
            for (int ks = 0; ks < 32; ++ks) {
                v16bf A = cat8(*(const v8bf*)&hpart[arow + ks * 32 + kofs],
                               *(const v8bf*)&hpart[arow + ks * 32 + kofs + 16]);
                v16bf Bm = *(const v16bf*)(bp + ((size_t)(32 + ks) * 32 + lane) * 16);
                acc = __builtin_amdgcn_wmma_f32_16x16x32_bf16(
                    false, A, false, Bm, (short)0, acc, false, false);
            }
            int n = nt * 16 + (lane & 15);
            int mb = mt * 16 + ((lane < 16) ? 0 : 8);
            float bv = bc[n];
#pragma unroll
            for (int v = 0; v < 8; ++v) {
                int row = mb + v;
                float c  = fast_tanh(acc[v] + bv);
                float u  = ubuf[row * D + n];
                float ho = hcur[row * D + n];
                float hn = (1.0f - u) * ho + u * c;
                hnxt[row * D + n] = hn;
                hidden[((size_t)row * S + t) * D + n] = hn;
            }
        }
        grid_barrier(cnt, gen, NBLK_GRU);
    }

    // ---- final carry: S even -> final state in buffer 0 ----
    if (blockIdx.x == 0) {
        const float* hf = h0buf;
        for (int i = tid; i < B * D; i += 256) {
            int b = i >> 10, d = i & (D - 1);
            float v = hf[i];
            if (d < DC) out[O_FCR + b * DC + d] = v;
            else        out[O_FCI + b * DC + (d - DC)] = v;
        }
    }
}

// ---------------------------------------------------------------------------
// Kernel 5: layernorm of hidden halves -> n_r, n_i (fp32 into d_out)
// ---------------------------------------------------------------------------
__global__ __launch_bounds__(256) void ln_kernel(
    const float* __restrict__ hidden,
    const float* __restrict__ sr, const float* __restrict__ br,
    const float* __restrict__ si, const float* __restrict__ bi,
    float* __restrict__ out) {

    const int row = blockIdx.x;                 // 0 .. B*S-1
    const float* x = hidden + (size_t)row * D;
    const int tid = threadIdx.x, lane = tid & 31, wid = tid >> 5;

    __shared__ float part[8][2];
    __shared__ float stat[2];

    for (int h = 0; h < 2; ++h) {
        float s = 0.f, s2 = 0.f;
        for (int j = tid; j < DC; j += 256) {
            float v = x[h * DC + j];
            s += v; s2 += v * v;
        }
#pragma unroll
        for (int o = 16; o > 0; o >>= 1) {
            s  += __shfl_down(s, o, 32);
            s2 += __shfl_down(s2, o, 32);
        }
        if (lane == 0) { part[wid][0] = s; part[wid][1] = s2; }
        __syncthreads();
        if (tid == 0) {
            float a = 0.f, b = 0.f;
#pragma unroll
            for (int i = 0; i < 8; ++i) { a += part[i][0]; b += part[i][1]; }
            float mean = a * (1.0f / DC);
            float var  = b * (1.0f / DC) - mean * mean;
            stat[0] = mean;
            stat[1] = rsqrtf(var + EPS);
        }
        __syncthreads();
        float mean = stat[0], rsq = stat[1];
        const float* sc = (h == 0) ? sr : si;
        const float* bs = (h == 0) ? br : bi;
        size_t obase = ((h == 0) ? O_NR : O_NI) + (size_t)row * DC;
        for (int j = tid; j < DC; j += 256) {
            float v = x[h * DC + j];
            out[obase + j] = (v - mean) * rsq * sc[j] + bs[j];
        }
        __syncthreads();
    }
}

// ---------------------------------------------------------------------------
// Kernel 6: logits = (n_r @ W_rc + b_rc)^2 + (n_i @ W_ic + b_ic)^2
// ---------------------------------------------------------------------------
__global__ __launch_bounds__(256) void logits_kernel(
    const float* __restrict__ out_ro,   // d_out (reads n_r/n_i regions)
    const __bf16* __restrict__ wo,
    const float* __restrict__ brc, const float* __restrict__ bic,
    float* __restrict__ out) {

    __shared__ __bf16 nr[64 * DC];      // 64 KB
    __shared__ __bf16 ni[64 * DC];      // 64 KB

    const int tid = threadIdx.x, lane = tid & 31, wid = tid >> 5;
    const int row0 = blockIdx.x * 64;

    for (int i = tid; i < 64 * DC; i += 256) {
        int r = i >> 9, k = i & (DC - 1);
        nr[i] = (__bf16)out_ro[O_NR + (size_t)(row0 + r) * DC + k];
        ni[i] = (__bf16)out_ro[O_NI + (size_t)(row0 + r) * DC + k];
    }
    __syncthreads();

    const int mt  = wid >> 1;           // 0..3 (16-row tile)
    const int ntb = (wid & 1) * 8;      // 8 column tiles per wave
    const int arow = (mt * 16 + (lane & 15)) * DC;
    const int kofs = (lane < 16) ? 0 : 8;

    for (int nt = ntb; nt < ntb + 8; ++nt) {
        v8f accR = {}, accI = {};
#pragma unroll
        for (int ks = 0; ks < 16; ++ks) {
            v16bf aR = cat8(*(const v8bf*)&nr[arow + ks * 32 + kofs],
                            *(const v8bf*)&nr[arow + ks * 32 + kofs + 16]);
            v16bf aI = cat8(*(const v8bf*)&ni[arow + ks * 32 + kofs],
                            *(const v8bf*)&ni[arow + ks * 32 + kofs + 16]);
            v16bf BR = *(const v16bf*)(wo + ((size_t)((0 * 16 + nt) * 16 + ks) * 32 + lane) * 16);
            v16bf BI = *(const v16bf*)(wo + ((size_t)((1 * 16 + nt) * 16 + ks) * 32 + lane) * 16);
            accR = __builtin_amdgcn_wmma_f32_16x16x32_bf16(
                false, aR, false, BR, (short)0, accR, false, false);
            accI = __builtin_amdgcn_wmma_f32_16x16x32_bf16(
                false, aI, false, BI, (short)0, accI, false, false);
        }
        int n  = nt * 16 + (lane & 15);
        int mb = mt * 16 + ((lane < 16) ? 0 : 8);
        float bR = brc[n], bI = bic[n];
#pragma unroll
        for (int v = 0; v < 8; ++v) {
            float xr = accR[v] + bR;
            float xi = accI[v] + bI;
            out[O_LOG + (size_t)(row0 + mb + v) * V + n] = xr * xr + xi * xi;
        }
    }
}

// ---------------------------------------------------------------------------
// Host launcher
// ---------------------------------------------------------------------------
extern "C" void kernel_launch(void* const* d_in, const int* in_sizes, int n_in,
                              void* d_out, int out_size, void* d_ws, size_t ws_size,
                              hipStream_t stream) {
    (void)in_sizes; (void)n_in; (void)out_size; (void)ws_size;

    const int*   tok  = (const int*)  d_in[0];
    const float* Er   = (const float*)d_in[1];
    const float* Ei   = (const float*)d_in[2];
    const float* Wr   = (const float*)d_in[3];
    const float* br   = (const float*)d_in[4];
    const float* Wu   = (const float*)d_in[5];
    const float* bu   = (const float*)d_in[6];
    const float* Wc   = (const float*)d_in[7];
    const float* bc   = (const float*)d_in[8];
    const float* lnrs = (const float*)d_in[9];
    const float* lnrb = (const float*)d_in[10];
    const float* lnis = (const float*)d_in[11];
    const float* lnib = (const float*)d_in[12];
    const float* Wrc  = (const float*)d_in[13];
    const float* brc  = (const float*)d_in[14];
    const float* Wic  = (const float*)d_in[15];
    const float* bic  = (const float*)d_in[16];

    char* ws = (char*)d_ws;
    unsigned* bar  = (unsigned*)(ws + WS_BAR);
    float*    hbuf = (float*)   (ws + WS_H);
    float*    rbuf = (float*)   (ws + WS_R);
    float*    ubuf = (float*)   (ws + WS_U);
    __bf16*   wp   = (__bf16*)  (ws + WS_WP);
    __bf16*   wo   = (__bf16*)  (ws + WS_WO);
    float*    hid  = (float*)   (ws + WS_HID);
    __bf16*   xs   = (__bf16*)  (ws + WS_XS);

    float* out = (float*)d_out;

    init_kernel<<<(2 * B * D + 255) / 256, 256, 0, stream>>>(bar, hbuf);
    embed_kernel<<<(int)(((size_t)B * S * D + 255) / 256), 256, 0, stream>>>(tok, Er, Ei, xs);
    pack_gates_kernel<<<(3 * 64 * 64 * 32 + 255) / 256, 256, 0, stream>>>(Wr, Wu, Wc, wp);
    pack_out_kernel<<<(2 * 16 * 16 * 32 + 255) / 256, 256, 0, stream>>>(Wrc, Wic, wo);
    gru_kernel<<<NBLK_GRU, 256, 0, stream>>>(xs, br, bu, bc, wp,
                                             hbuf, rbuf, ubuf, hid, bar, out);
    ln_kernel<<<B * S, 256, 0, stream>>>(hid, lnrs, lnrb, lnis, lnib, out);
    logits_kernel<<<(B * S) / 64, 256, 0, stream>>>(out, wo, brc, bic, out);
}
